// MoE_76759655514826
// MI455X (gfx1250) — matrix operand order repaired and checked by reading, not compiled, not run on hardware
//
#include <hip/hip_runtime.h>
#include <math.h>

// ---------------------------------------------------------------------------
// MoE (T=16384, D=512, F=2048, E=8, top-2) for gfx1250.
// bf16 WMMA (v_wmma_f32_16x16x32_bf16) for both FFN GEMMs, f32 router.
// ---------------------------------------------------------------------------

typedef __bf16 bf16;
typedef __attribute__((ext_vector_type(16))) __bf16 v16bf;
typedef __attribute__((ext_vector_type(8)))  __bf16 v8bf;
typedef __attribute__((ext_vector_type(8)))  float  v8f;

#define T_TOK 16384
#define DM    512
#define FF    2048
#define NE    8
#define XP    (DM + 8)   // LDS pitch for X tile (bf16 elements)
#define HP    (FF + 8)   // LDS pitch for H tile (bf16 elements)
#define RT    256        // tokens per router block
#define RPITCH 65        // router LDS pitch (floats): bank = (row + d) % 64

// --------------------------- zero output + counters -------------------------
__global__ void k_zero(float* __restrict__ out, int* __restrict__ counts) {
  size_t i = (size_t)blockIdx.x * blockDim.x + threadIdx.x;
  size_t stride = (size_t)gridDim.x * blockDim.x;
  size_t n = (size_t)T_TOK * DM;
  for (size_t j = i; j < n; j += stride) out[j] = 0.0f;
  if (i < NE) counts[i] = 0;
}

// --------------------------- f32 -> bf16 conversion -------------------------
// T*DM == NE*FF*DM == NE*DM*FF == 8388608, so one loop converts all three.
__global__ void k_convert(const float* __restrict__ x,
                          const float* __restrict__ w1,
                          const float* __restrict__ w2,
                          bf16* __restrict__ xb,
                          bf16* __restrict__ w1b,
                          bf16* __restrict__ w2b) {
  size_t n = (size_t)T_TOK * DM;
  size_t stride = (size_t)gridDim.x * blockDim.x;
  for (size_t j = (size_t)blockIdx.x * blockDim.x + threadIdx.x; j < n; j += stride) {
    xb[j]  = (bf16)x[j];
    w1b[j] = (bf16)w1[j];
    w2b[j] = (bf16)w2[j];
  }
}

// --------------------------- router (exact f32) -----------------------------
// 256 tokens per block; x tiled through LDS in 64-wide d-chunks with fully
// coalesced float4 global reads.  Pitch 65 -> conflict-free compute reads.
__global__ __launch_bounds__(256) void k_router(const float* __restrict__ x,
                                                const float* __restrict__ gate,
                                                float* __restrict__ logits_out,
                                                float* __restrict__ ent,
                                                int*   __restrict__ counts,
                                                int*   __restrict__ perm,
                                                float* __restrict__ pw) {
  __shared__ float g[NE * DM];        // 16 KB
  __shared__ float xt[RT * RPITCH];   // 65 KB

  const int tid = threadIdx.x;
  for (int i = tid; i < NE * DM; i += blockDim.x) g[i] = gate[i];

  const int t0 = blockIdx.x * RT;
  float acc[NE];
#pragma unroll
  for (int e = 0; e < NE; ++e) acc[e] = 0.0f;

  const int seg = tid & 15;           // 16 lanes cover one 64-float row chunk
  for (int dc = 0; dc < DM; dc += 64) {
    __syncthreads();                  // xt reuse barrier (also covers g on dc==0)
    for (int r = tid >> 4; r < RT; r += 16) {
      float4 v = *(const float4*)(x + (size_t)(t0 + r) * DM + dc + seg * 4);
      // scalar LDS stores (pitch 65 is not 16B-aligned per row)
      xt[r * RPITCH + seg * 4 + 0] = v.x;
      xt[r * RPITCH + seg * 4 + 1] = v.y;
      xt[r * RPITCH + seg * 4 + 2] = v.z;
      xt[r * RPITCH + seg * 4 + 3] = v.w;
    }
    __syncthreads();
    const float* xr = &xt[tid * RPITCH];
    for (int d = 0; d < 64; ++d) {
      float xv = xr[d];
#pragma unroll
      for (int e = 0; e < NE; ++e) acc[e] = fmaf(xv, g[e * DM + dc + d], acc[e]);
    }
  }

  const int t = t0 + tid;

  // softmax (stable)
  float mx = acc[0];
#pragma unroll
  for (int e = 1; e < NE; ++e) mx = fmaxf(mx, acc[e]);
  float p[NE];
  float s = 0.0f;
#pragma unroll
  for (int e = 0; e < NE; ++e) { p[e] = expf(acc[e] - mx); s += p[e]; }
  float inv = 1.0f / s;

  float ent_t = 0.0f;
#pragma unroll
  for (int e = 0; e < NE; ++e) {
    p[e] *= inv;
    logits_out[(size_t)t * NE + e] = acc[e];
    ent_t -= p[e] * logf(fmaxf(p[e], 1e-6f));
  }
  ent[t] = ent_t;

  // top-2 (ties -> lower index first, matching lax.top_k)
  int   b0 = 0, b1 = -1;
  float v0 = -1.0f, v1 = -1.0f;
#pragma unroll
  for (int e = 0; e < NE; ++e) {
    if (p[e] > v0)      { v1 = v0; b1 = b0; v0 = p[e]; b0 = e; }
    else if (p[e] > v1) { v1 = p[e]; b1 = e; }
  }
  float wsum = v0 + v1;
  float rw0 = v0 / wsum, rw1 = v1 / wsum;

  int pos0 = atomicAdd(&counts[b0], 1);
  perm[(size_t)b0 * T_TOK + pos0] = t;
  pw[(size_t)b0 * T_TOK + pos0]   = rw0;
  int pos1 = atomicAdd(&counts[b1], 1);
  perm[(size_t)b1 * T_TOK + pos1] = t;
  pw[(size_t)b1 * T_TOK + pos1]   = rw1;
}

// --------------------------- fused expert FFN (WMMA) ------------------------
// One block: expert e, tile of 16 token-slots.  8 waves, 256 threads.
// Stage A: H[16,2048] = gelu(X[16,512] @ W1[e]^T)   (bf16 WMMA, f32 acc)
// Stage B: OUT[16,512] = H @ W2[e]^T, weighted atomic-add into output.
// Two N-tiles per A fragment (dual accumulators) to halve LDS traffic/WMMA.
__global__ __launch_bounds__(256) void k_ffn(const bf16* __restrict__ xb,
                                             const bf16* __restrict__ w1b,
                                             const bf16* __restrict__ w2b,
                                             const int*  __restrict__ counts,
                                             const int*  __restrict__ perm,
                                             const float* __restrict__ pw,
                                             float* __restrict__ out) {
  __shared__ bf16  xs[16 * XP];
  __shared__ bf16  hs[16 * HP];
  __shared__ int   toks[16];
  __shared__ float wgts[16];

  const int e    = blockIdx.x >> 10;    // 1024 tiles per expert
  const int tile = blockIdx.x & 1023;
  const int cnt  = counts[e];
  if (tile * 16 >= cnt) return;         // block-uniform early exit

  const int tid  = threadIdx.x;
  const int lane = tid & 31;
  const int wave = tid >> 5;
  const int half = lane >> 4;           // K-range selector: k = 16*half + s (contiguous)
  const int l16  = lane & 15;           // M (A) / N (B,C,D) index

  if (tid < 16) {
    int slot = tile * 16 + tid;
    if (slot < cnt) {
      toks[tid] = perm[(size_t)e * T_TOK + slot];
      wgts[tid] = pw[(size_t)e * T_TOK + slot];
    } else {
      toks[tid] = -1;
      wgts[tid] = 0.0f;
    }
  }
  __syncthreads();

  // ---- stage gathered X tile into LDS (row-major, pitch XP) ----
  {
    int row = tid >> 4;                 // 0..15
    int seg = tid & 15;                 // 32 elements each
    int tok = toks[row];
    bf16* dst = xs + row * XP + seg * 32;
    if (tok >= 0) {
      const bf16* src = xb + (size_t)tok * DM + seg * 32;
      *(v8bf*)(dst +  0) = *(const v8bf*)(src +  0);
      *(v8bf*)(dst +  8) = *(const v8bf*)(src +  8);
      *(v8bf*)(dst + 16) = *(const v8bf*)(src + 16);
      *(v8bf*)(dst + 24) = *(const v8bf*)(src + 24);
    } else {
      v8bf z = {};
      *(v8bf*)(dst +  0) = z;
      *(v8bf*)(dst +  8) = z;
      *(v8bf*)(dst + 16) = z;
      *(v8bf*)(dst + 24) = z;
    }
  }
  __syncthreads();

  // ---- Stage A: H = gelu(X @ W1^T); wave owns 16 contiguous f-tiles ----
  {
    const bf16* arow = xs + l16 * XP + half * 16;
    const int ftbase = wave * 16;
    for (int j = 0; j < 16; j += 2) {
      const int ft0 = ftbase + j;
      v8f acc0 = {}, acc1 = {};
      const bf16* b0row = w1b + ((size_t)e * FF + ft0 * 16 + l16) * DM + half * 16;
      const bf16* b1row = b0row + (size_t)16 * DM;
      if (j + 2 < 16) __builtin_prefetch(b1row + (size_t)16 * DM, 0, 1);
#pragma unroll 4
      for (int kb = 0; kb < DM / 32; ++kb) {
        v8bf alo = *(const v8bf*)(arow + kb * 32);
        v8bf ahi = *(const v8bf*)(arow + kb * 32 + 8);
        v16bf a = __builtin_shufflevector(alo, ahi, 0,1,2,3,4,5,6,7,8,9,10,11,12,13,14,15);
        v16bf b0 = *(const v16bf*)(b0row + kb * 32);
        v16bf b1 = *(const v16bf*)(b1row + kb * 32);
        acc0 = __builtin_amdgcn_wmma_f32_16x16x32_bf16(false, a, false, b0,
                                                       (short)0, acc0, false, false);
        acc1 = __builtin_amdgcn_wmma_f32_16x16x32_bf16(false, a, false, b1,
                                                       (short)0, acc1, false, false);
      }
      // exact GELU, convert to bf16, store D tiles into H (row m, col f)
#pragma unroll
      for (int r = 0; r < 8; ++r) {
        int m = r + 8 * half;
        float va = acc0[r];
        float vb = acc1[r];
        float ga = 0.5f * va * (1.0f + erff(va * 0.70710678118654752440f));
        float gb = 0.5f * vb * (1.0f + erff(vb * 0.70710678118654752440f));
        hs[m * HP + ft0 * 16 + l16]      = (bf16)ga;
        hs[m * HP + (ft0 + 1) * 16 + l16] = (bf16)gb;
      }
    }
  }
  __syncthreads();

  // ---- Stage B: OUT = H @ W2^T; wave owns 4 contiguous d-tiles ----
  {
    const bf16* arow = hs + l16 * HP + half * 16;
    const int dtbase = wave * 4;
    for (int j = 0; j < 4; j += 2) {
      const int dt0 = dtbase + j;
      v8f acc0 = {}, acc1 = {};
      const bf16* b0row = w2b + ((size_t)e * DM + dt0 * 16 + l16) * FF + half * 16;
      const bf16* b1row = b0row + (size_t)16 * FF;
      if (j + 2 < 4) __builtin_prefetch(b1row + (size_t)16 * FF, 0, 1);
#pragma unroll 4
      for (int kb = 0; kb < FF / 32; ++kb) {
        v8bf alo = *(const v8bf*)(arow + kb * 32);
        v8bf ahi = *(const v8bf*)(arow + kb * 32 + 8);
        v16bf a = __builtin_shufflevector(alo, ahi, 0,1,2,3,4,5,6,7,8,9,10,11,12,13,14,15);
        v16bf b0 = *(const v16bf*)(b0row + kb * 32);
        v16bf b1 = *(const v16bf*)(b1row + kb * 32);
        acc0 = __builtin_amdgcn_wmma_f32_16x16x32_bf16(false, a, false, b0,
                                                       (short)0, acc0, false, false);
        acc1 = __builtin_amdgcn_wmma_f32_16x16x32_bf16(false, a, false, b1,
                                                       (short)0, acc1, false, false);
      }
#pragma unroll
      for (int r = 0; r < 8; ++r) {
        int m = r + 8 * half;
        int tok = toks[m];
        if (tok >= 0) {
          float w = wgts[m];
          atomicAdd(out + (size_t)tok * DM + dt0 * 16 + l16,        acc0[r] * w);
          atomicAdd(out + (size_t)tok * DM + (dt0 + 1) * 16 + l16,  acc1[r] * w);
        }
      }
    }
  }
}

// --------------------------- deterministic entropy mean ---------------------
__global__ void k_reduce(const float* __restrict__ ent, float* __restrict__ out_scalar) {
  __shared__ float s[256];
  float acc = 0.0f;
  int base = threadIdx.x * (T_TOK / 256);
  for (int i = 0; i < T_TOK / 256; ++i) acc += ent[base + i];
  s[threadIdx.x] = acc;
  __syncthreads();
  for (int off = 128; off > 0; off >>= 1) {
    if (threadIdx.x < off) s[threadIdx.x] += s[threadIdx.x + off];
    __syncthreads();
  }
  if (threadIdx.x == 0) *out_scalar = s[0] / (float)T_TOK;
}

// ---------------------------------------------------------------------------
extern "C" void kernel_launch(void* const* d_in, const int* in_sizes, int n_in,
                              void* d_out, int out_size, void* d_ws, size_t ws_size,
                              hipStream_t stream) {
  (void)in_sizes; (void)n_in; (void)out_size; (void)ws_size;

  const float* x    = (const float*)d_in[0];   // [T, DM]
  const float* gate = (const float*)d_in[1];   // [NE, DM]
  const float* w1   = (const float*)d_in[2];   // [NE, FF, DM]
  const float* w2   = (const float*)d_in[3];   // [NE, DM, FF]

  float* out      = (float*)d_out;                    // [T*DM]
  float* logits   = out + (size_t)T_TOK * DM;         // [T*NE]
  float* ent_loss = logits + (size_t)T_TOK * NE;      // [1]

  char* ws = (char*)d_ws;
  bf16* xb   = (bf16*)ws;  ws += (size_t)T_TOK * DM * 2;   // 16 MB
  bf16* w1b  = (bf16*)ws;  ws += (size_t)NE * FF * DM * 2; // 16 MB
  bf16* w2b  = (bf16*)ws;  ws += (size_t)NE * DM * FF * 2; // 16 MB
  int*  counts = (int*)ws; ws += 256;
  int*  perm   = (int*)ws; ws += (size_t)NE * T_TOK * 4;   // 512 KB
  float* pwt   = (float*)ws; ws += (size_t)NE * T_TOK * 4; // 512 KB
  float* ent   = (float*)ws;                               // 64 KB

  k_zero   <<<2048, 256, 0, stream>>>(out, counts);
  k_convert<<<4096, 256, 0, stream>>>(x, w1, w2, xb, w1b, w2b);
  k_router <<<T_TOK / RT, 256, 0, stream>>>(x, gate, logits, ent, counts, perm, pwt);
  k_ffn    <<<NE * (T_TOK / 16), 256, 0, stream>>>(xb, w1b, w2b, counts, perm, pwt, out);
  k_reduce <<<1, 256, 0, stream>>>(ent, ent_loss);
}
